// GCN_residual_11914239279203
// MI455X (gfx1250) — matrix operand order repaired and checked by reading, not compiled, not run on hardware
//
#include <hip/hip_runtime.h>

#define D 128

typedef __attribute__((ext_vector_type(16))) _Float16 v16h;
typedef __attribute__((ext_vector_type(8)))  float    v8f;

// ---------------- utility kernels ----------------

__global__ __launch_bounds__(256) void zero_f32v4(float* __restrict__ p, long long n4) {
  long long i = (long long)blockIdx.x * blockDim.x + threadIdx.x;
  if (i < n4) ((float4*)p)[i] = make_float4(0.f, 0.f, 0.f, 0.f);
}

__global__ __launch_bounds__(256) void edge_deg(const int* __restrict__ col,
                                                const float* __restrict__ w,
                                                float* __restrict__ deg, int E) {
  int e = blockIdx.x * blockDim.x + threadIdx.x;
  if (e < E) unsafeAtomicAdd(&deg[col[e]], w[e]);
}

__global__ __launch_bounds__(256) void deg_to_dis(float* __restrict__ deg, int n) {
  int i = blockIdx.x * blockDim.x + threadIdx.x;
  if (i < n) {
    float d = deg[i];
    deg[i] = (d > 0.f) ? rsqrtf(d) : 0.f;   // in-place: deg -> D^{-1/2}
  }
}

__global__ __launch_bounds__(256) void edge_norm(const int* __restrict__ row,
                                                 const int* __restrict__ col,
                                                 const float* __restrict__ w,
                                                 const float* __restrict__ dis,
                                                 float* __restrict__ nrm, int E) {
  int e = blockIdx.x * blockDim.x + threadIdx.x;
  if (e < E) nrm[e] = dis[row[e]] * w[e] * dis[col[e]];
}

// Pack W (D x D f32, row-major, W[k][n]) into this code's B-fragment layout (k contiguous per lane half):
// index = ((nt*4 + kt)*32 + lane)*16 + i  holds  W[kt*32 + (lane>>4)*16 + i][nt*16 + (lane&15)]
__global__ __launch_bounds__(256) void pack_w(const float* __restrict__ W, _Float16* __restrict__ Wp) {
  int idx = blockIdx.x * blockDim.x + threadIdx.x;   // 0 .. 8*4*32*16-1 = 16383
  if (idx >= 8 * 4 * 32 * 16) return;
  int i    = idx & 15;
  int lane = (idx >> 4) & 31;
  int kt   = (idx >> 9) & 3;
  int nt   = idx >> 11;
  int n = nt * 16 + (lane & 15);
  int k = kt * 32 + (lane >> 4) * 16 + i;
  Wp[idx] = (_Float16)W[k * D + n];
}

// ---------------- WMMA GEMM: Y[N x 128] = X[N x 128] * W ----------------
// grid.x = N/16 row strips; 8 waves per block, wave w owns column tile w.
__global__ __launch_bounds__(256) void gemm128_wmma(const float* __restrict__ X,
                                                    const _Float16* __restrict__ Wp,
                                                    float* __restrict__ Y) {
  const int wave  = threadIdx.x >> 5;   // 0..7 -> n-tile
  const int lane  = threadIdx.x & 31;
  const int m0    = blockIdx.x << 4;
  const int khalf = lane >> 4;          // 0 or 1
  const int mr    = lane & 15;

  const float* xrow = X + (long long)(m0 + mr) * D;
  const _Float16* wb = Wp + ((long long)wave * 4) * 32 * 16;

  v8f c = {};
#pragma unroll
  for (int kt = 0; kt < 4; ++kt) {
    // A fragment (16x32 f16): halves 0..7 -> K = kt*32 + khalf*16 + i,
    //                         halves 8..15 -> K = kt*32 + khalf*16 + 8 + i
    v16h a;
    const float* p0 = xrow + kt * 32 + khalf * 16;
#pragma unroll
    for (int i = 0; i < 8; ++i) a[i] = (_Float16)p0[i];
    const float* p1 = p0 + 8;
#pragma unroll
    for (int i = 0; i < 8; ++i) a[8 + i] = (_Float16)p1[i];

    // B fragment: pre-packed, one 32B load
    v16h b = *(const v16h*)(wb + ((long long)kt * 32 + lane) * 16);

    c = __builtin_amdgcn_wmma_f32_16x16x32_f16(false, a, false, b, (short)0, c, false, false);
  }

  // C/D layout: VGPR r -> M = m0 + r + 8*khalf, N = wave*16 + (lane&15)
  const int row = m0 + khalf * 8;
  const int col = (wave << 4) + mr;
  float* yp = Y + (long long)row * D + col;
#pragma unroll
  for (int r = 0; r < 8; ++r) yp[r * D] = c[r];
}

// ---------------- scatter-add: acc[col[e]] += xl[row[e]] * nrm[e] ----------------
// 32 lanes per edge, 4 features per lane (float4 gather + 4 fp32 hardware atomics)
__global__ __launch_bounds__(256) void scatter_edges(const float* __restrict__ xl,
                                                     const int* __restrict__ row,
                                                     const int* __restrict__ col,
                                                     const float* __restrict__ nrm,
                                                     float* __restrict__ acc, int E) {
  long long t = (long long)blockIdx.x * blockDim.x + threadIdx.x;
  int e = (int)(t >> 5);
  if (e >= E) return;
  int d = (int)(t & 31) << 2;
  int r = row[e], cc = col[e];
  float s = nrm[e];
  const float4 v = *(const float4*)(xl + (long long)r * D + d);
  float* dst = acc + (long long)cc * D + d;
  unsafeAtomicAdd(dst + 0, v.x * s);
  unsafeAtomicAdd(dst + 1, v.y * s);
  unsafeAtomicAdd(dst + 2, v.z * s);
  unsafeAtomicAdd(dst + 3, v.w * s);
}

// ---------------- epilogues ----------------

__global__ __launch_bounds__(256) void bias_relu(const float* __restrict__ acc,
                                                 const float* __restrict__ bias,
                                                 float* __restrict__ out, long long n4) {
  long long i = (long long)blockIdx.x * blockDim.x + threadIdx.x;
  if (i >= n4) return;
  float4 v = ((const float4*)acc)[i];
  int cb = (int)((i << 2) & (D - 1));
  v.x = fmaxf(v.x + bias[cb + 0], 0.f);
  v.y = fmaxf(v.y + bias[cb + 1], 0.f);
  v.z = fmaxf(v.z + bias[cb + 2], 0.f);
  v.w = fmaxf(v.w + bias[cb + 3], 0.f);
  ((float4*)out)[i] = v;
}

__global__ __launch_bounds__(256) void bias_relu_residual(const float* __restrict__ acc,
                                                          const float* __restrict__ bias,
                                                          const float* __restrict__ xres,
                                                          float* __restrict__ out, long long n4) {
  long long i = (long long)blockIdx.x * blockDim.x + threadIdx.x;
  if (i >= n4) return;
  float4 v = ((const float4*)acc)[i];
  float4 x = ((const float4*)xres)[i];
  int cb = (int)((i << 2) & (D - 1));
  v.x = fmaxf(v.x + bias[cb + 0], 0.f) + x.x;
  v.y = fmaxf(v.y + bias[cb + 1], 0.f) + x.y;
  v.z = fmaxf(v.z + bias[cb + 2], 0.f) + x.z;
  v.w = fmaxf(v.w + bias[cb + 3], 0.f) + x.w;
  ((float4*)out)[i] = v;
}

// ---------------- launch ----------------

extern "C" void kernel_launch(void* const* d_in, const int* in_sizes, int n_in,
                              void* d_out, int out_size, void* d_ws, size_t ws_size,
                              hipStream_t stream) {
  const float* x  = (const float*)d_in[0];
  const int*   adj = (const int*)d_in[1];          // (2,E) int32, row-major
  const float* w  = (const float*)d_in[2];
  const float* W1 = (const float*)d_in[3];
  const float* b1 = (const float*)d_in[4];
  const float* W2 = (const float*)d_in[5];
  const float* b2 = (const float*)d_in[6];

  const int Nn = in_sizes[0] / D;      // 50000
  const int E  = in_sizes[2];          // 600000
  const int* row = adj;                // source
  const int* col = adj + E;            // target

  const long long ND  = (long long)Nn * D;
  const long long ND4 = ND >> 2;

  // workspace layout
  float* deg  = (float*)d_ws;                  // N   (becomes dis in place)
  float* nrm  = deg + Nn;                      // E
  float* bufA = nrm + E;                       // N*D
  float* bufB = bufA + ND;                     // N*D
  _Float16* Wp1 = (_Float16*)(bufB + ND);      // 16384 halves
  _Float16* Wp2 = Wp1 + 16384;                 // 16384 halves

  const int TB = 256;
  dim3 blk(TB);
  int gE    = (E + TB - 1) / TB;
  int gN    = (Nn + TB - 1) / TB;
  long long gND4 = (ND4 + TB - 1) / TB;
  long long gSc  = (((long long)E << 5) + TB - 1) / TB;   // 32 lanes per edge
  int gN4   = ((Nn >> 2) + TB - 1) / TB;
  int gGemm = Nn / 16;                                     // 3125

  // normalization: deg -> dis -> per-edge norm
  zero_f32v4<<<gN4, blk, 0, stream>>>(deg, (long long)(Nn >> 2));
  edge_deg<<<gE, blk, 0, stream>>>(col, w, deg, E);
  deg_to_dis<<<gN, blk, 0, stream>>>(deg, Nn);
  edge_norm<<<gE, blk, 0, stream>>>(row, col, w, deg, nrm, E);

  // pack weights into WMMA B-fragment layout (f16)
  pack_w<<<64, blk, 0, stream>>>(W1, Wp1);
  pack_w<<<64, blk, 0, stream>>>(W2, Wp2);

  // ---- layer 1 ----
  gemm128_wmma<<<gGemm, blk, 0, stream>>>(x, Wp1, bufA);            // xl1 = x @ W1
  zero_f32v4<<<(int)gND4, blk, 0, stream>>>(bufB, ND4);             // acc1 = 0
  scatter_edges<<<(int)gSc, blk, 0, stream>>>(bufA, row, col, nrm, bufB, E);
  bias_relu<<<(int)gND4, blk, 0, stream>>>(bufB, b1, bufA, ND4);    // h1 = relu(acc1+b1)

  // ---- layer 2 ----
  gemm128_wmma<<<gGemm, blk, 0, stream>>>(bufA, Wp2, bufB);         // xl2 = h1 @ W2
  zero_f32v4<<<(int)gND4, blk, 0, stream>>>(bufA, ND4);             // acc2 = 0
  scatter_edges<<<(int)gSc, blk, 0, stream>>>(bufB, row, col, nrm, bufA, E);
  bias_relu_residual<<<(int)gND4, blk, 0, stream>>>(bufA, b2, x, (float*)d_out, ND4);
}
